// MessagePassingConvolution_50843822850738
// MI455X (gfx1250) — compile-verified, matrix operand order checked by reading, not run on hardware
//
#include <hip/hip_runtime.h>

typedef __attribute__((ext_vector_type(16))) _Float16 v16h;
typedef __attribute__((ext_vector_type(8)))  float    v8f;

#define CH 64
#define RD 8

// K index pattern for 16-bit A/B WMMA fragments (wave32):
// element e in v16h, lane in [0,32): K = (e&7) + ((e>>3)<<4) + ((lane>>4)<<3)
__device__ __forceinline__ int kpat(int e, int lane) {
  return (e & 7) + ((e >> 3) << 4) + ((lane >> 4) << 3);
}

// Load a pre-swizzled B fragment: layout [ntile][kfrag][lane][16 halves], contiguous per lane.
__device__ __forceinline__ v16h ldb(const _Float16* __restrict__ p, int nt, int kf, int KF, int lane) {
  return *(const v16h*)(p + ((size_t)((nt * KF + kf) * 32 + lane) << 4));
}

__device__ __forceinline__ float silu(float x) {
  return x / (1.0f + __expf(-x));
}

__device__ __forceinline__ v8f wmma16(v16h a, v16h b, v8f c) {
  return __builtin_amdgcn_wmma_f32_16x16x32_f16(false, a, false, b, (short)0, c, false, false);
}

__global__ void zero_kernel(float* __restrict__ out, int n) {
  int i = blockIdx.x * blockDim.x + threadIdx.x;
  int stride = gridDim.x * blockDim.x;
  for (; i < n; i += stride) out[i] = 0.0f;
}

// Convert one f32 weight matrix [Krows x N] into swizzled f16 B-fragments, scale folded in.
__device__ __forceinline__ void pack_one(const float* __restrict__ w, int Krows, int N, int KF,
                                         float scale, _Float16* __restrict__ dst, int li) {
  int e    = li & 15;
  int lane = (li >> 4) & 31;
  int fk   = li >> 9;            // combined (nt*KF + kf)
  int nt   = fk / KF;
  int kf   = fk - nt * KF;
  int K    = kf * 32 + kpat(e, lane);
  int n    = nt * 16 + (lane & 15);
  float v  = (K < Krows) ? w[K * N + n] * scale : 0.0f;
  dst[li]  = (_Float16)v;
}

__global__ void pack_weights(const float* __restrict__ w1, const float* __restrict__ w2,
                             const float* __restrict__ w3, const float* __restrict__ w4,
                             _Float16* __restrict__ ws) {
  int t = blockIdx.x * blockDim.x + threadIdx.x;
  // w1: K=8 (padded to 32), N=64, KF=1 -> 2048 elems ; scale 1/sqrt(8)
  // w2/w3: K=64, N=64, KF=2 -> 4096 each ; scale 1/8
  // w4: K=64, N=256, KF=2 -> 16384 ; scale 1/8 * 1/sqrt(16)
  if (t < 2048)        pack_one(w1,  8,  64, 1, 0.35355339059327373f, ws,          t);
  else if (t < 6144)   pack_one(w2, 64,  64, 2, 0.125f,               ws + 2048,   t - 2048);
  else if (t < 10240)  pack_one(w3, 64,  64, 2, 0.125f,               ws + 6144,   t - 6144);
  else if (t < 26624)  pack_one(w4, 64, 256, 2, 0.03125f,             ws + 10240,  t - 10240);
}

template <int L>
__device__ __forceinline__ void emit_msgs(const float* __restrict__ sb, const float* __restrict__ mb,
                                          int half, float x, float y, float z,
                                          float* __restrict__ op) {
  constexpr int dim  = (L == 0) ? 1 : (L == 1 ? 3 : (L == 2 ? 5 : 7));
  constexpr int base = (L == 0) ? 0 : (L == 1 ? 64 : (L == 2 ? 256 : 576));
  float yv[dim];
  if constexpr (L == 0) {
    yv[0] = 1.0f;
  } else if constexpr (L == 1) {
    const float s3 = 1.7320508075688772f;
    yv[0] = s3 * y; yv[1] = s3 * z; yv[2] = s3 * x;
  } else if constexpr (L == 2) {
    const float s5 = 2.23606797749979f, s15 = 3.872983346207417f;
    yv[0] = s15 * x * y;
    yv[1] = s15 * y * z;
    yv[2] = 0.5f * s5 * (3.0f * z * z - 1.0f);
    yv[3] = s15 * x * z;
    yv[4] = 0.5f * s15 * (x * x - y * y);
  } else {
    const float c33 = 2.091650066335189f, c32 = 10.246950765959598f;
    const float c31 = 1.6201851746019651f, c30 = 1.3228756555322954f;
    yv[0] = c33 * y * (3.0f * x * x - y * y);
    yv[1] = c32 * x * y * z;
    yv[2] = c31 * y * (5.0f * z * z - 1.0f);
    yv[3] = c30 * z * (5.0f * z * z - 3.0f);
    yv[4] = c31 * x * (5.0f * z * z - 1.0f);
    yv[5] = 0.5f * c32 * z * (x * x - y * y);
    yv[6] = c33 * x * (x * x - 3.0f * y * y);
  }
  for (int ci = 0; ci < 32; ++ci) {
    int c = half * 32 + ci;
    float sc = sb[c] * mb[c];
    #pragma unroll
    for (int j = 0; j < dim; ++j)
      __hip_atomic_fetch_add(&op[base + c * dim + j], sc * yv[j],
                             __ATOMIC_RELAXED, __HIP_MEMORY_SCOPE_AGENT);
  }
}

__global__ void __launch_bounds__(128, 1) mace_conv_kernel(
    const float* __restrict__ vectors,
    const float* __restrict__ node_feats,
    const float* __restrict__ radial,
    const int*   __restrict__ senders,
    const int*   __restrict__ receivers,
    const _Float16* __restrict__ wsp,
    float* __restrict__ out,
    int E) {
  __shared__ float    sbuf[4][16 * CH];   // gathered sender feats   (f32)
  __shared__ _Float16 actA[4][16 * CH];   // activation ping         (f16)
  __shared__ _Float16 actB[4][16 * CH];   // activation pong         (f16)
  __shared__ float    mixb[4][16 * CH];   // mix chunk for current l (f32)

  const int wv = threadIdx.x >> 5;
  const int tl = threadIdx.x & 31;
  const int tile = blockIdx.x * 4 + wv;
  const int nTiles = (E + 15) >> 4;
  if (tile >= nTiles) return;            // wave-uniform: EXEC stays all-ones for WMMA
  const int e0 = tile << 4;

  const _Float16* w1p = wsp;
  const _Float16* w2p = wsp + 2048;
  const _Float16* w3p = wsp + 6144;
  const _Float16* w4p = wsp + 10240;

  // CDNA5 prefetch of the next tile's radial embedding (global_prefetch_b8)
  if (e0 + 16 < E) __builtin_prefetch(&radial[(size_t)(e0 + 16) * RD], 0, 1);

  // ---- gather sender features into LDS (16 edges x 64 ch) ----
  for (int e = 0; e < 16; ++e) {
    int eg = e0 + e; if (eg >= E) eg = E - 1;
    int snd = senders[eg];
    sbuf[wv][e * CH + tl]      = node_feats[(size_t)snd * CH + tl];
    sbuf[wv][e * CH + tl + 32] = node_feats[(size_t)snd * CH + tl + 32];
  }

  // ---- layer 1: silu(R[16x8] @ w1 / sqrt(8)) via zero-padded K=32 WMMA ----
  {
    int erow = e0 + (tl & 15); if (erow >= E) erow = E - 1;
    v16h a;
    #pragma unroll
    for (int e = 0; e < 16; ++e) {
      int k = kpat(e, tl);
      a[e] = (k < RD) ? (_Float16)radial[(size_t)erow * RD + k] : (_Float16)0.0f;
    }
    #pragma unroll
    for (int nt = 0; nt < 4; ++nt) {
      v8f acc = {};
      acc = wmma16(a, ldb(w1p, nt, 0, 1, tl), acc);
      #pragma unroll
      for (int r = 0; r < 8; ++r)
        actA[wv][(r + ((tl >> 4) << 3)) * CH + nt * 16 + (tl & 15)] = (_Float16)silu(acc[r]);
    }
  }

  // ---- layer 2: silu(h @ w2 / 8) ----
  {
    v16h a0, a1;
    const int row = (tl & 15) * CH;
    #pragma unroll
    for (int e = 0; e < 16; ++e) {
      a0[e] = actA[wv][row + kpat(e, tl)];
      a1[e] = actA[wv][row + 32 + kpat(e, tl)];
    }
    #pragma unroll
    for (int nt = 0; nt < 4; ++nt) {
      v8f acc = {};
      acc = wmma16(a0, ldb(w2p, nt, 0, 2, tl), acc);
      acc = wmma16(a1, ldb(w2p, nt, 1, 2, tl), acc);
      #pragma unroll
      for (int r = 0; r < 8; ++r)
        actB[wv][(r + ((tl >> 4) << 3)) * CH + nt * 16 + (tl & 15)] = (_Float16)silu(acc[r]);
    }
  }

  // ---- layer 3: silu(h @ w3 / 8) ----
  {
    v16h a0, a1;
    const int row = (tl & 15) * CH;
    #pragma unroll
    for (int e = 0; e < 16; ++e) {
      a0[e] = actB[wv][row + kpat(e, tl)];
      a1[e] = actB[wv][row + 32 + kpat(e, tl)];
    }
    #pragma unroll
    for (int nt = 0; nt < 4; ++nt) {
      v8f acc = {};
      acc = wmma16(a0, ldb(w3p, nt, 0, 2, tl), acc);
      acc = wmma16(a1, ldb(w3p, nt, 1, 2, tl), acc);
      #pragma unroll
      for (int r = 0; r < 8; ++r)
        actA[wv][(r + ((tl >> 4) << 3)) * CH + nt * 16 + (tl & 15)] = (_Float16)silu(acc[r]);
    }
  }

  // ---- per-lane edge geometry for the message phase ----
  const int  me   = tl >> 1;        // edge 0..15 within tile
  const int  half = tl & 1;         // channel half 0..1
  int eg = e0 + me;
  const bool valid = (eg < E);
  if (!valid) eg = E - 1;
  float vx = vectors[(size_t)eg * 3 + 0];
  float vy = vectors[(size_t)eg * 3 + 1];
  float vz = vectors[(size_t)eg * 3 + 2];
  float inv = rsqrtf(vx * vx + vy * vy + vz * vz + 1e-24f);
  vx *= inv; vy *= inv; vz *= inv;
  const int recv = receivers[eg];
  float* op = out + (size_t)recv * (16 * CH);
  const float* sb = &sbuf[wv][me * CH];
  const float* mb = &mixb[wv][me * CH];

  // ---- layer 4 (mix = h @ w4 * 1/32) interleaved with message scatter per irrep l ----
  v16h a0, a1;
  {
    const int row = (tl & 15) * CH;
    #pragma unroll
    for (int e = 0; e < 16; ++e) {
      a0[e] = actA[wv][row + kpat(e, tl)];
      a1[e] = actA[wv][row + 32 + kpat(e, tl)];
    }
  }
  #pragma unroll
  for (int l = 0; l < 4; ++l) {
    #pragma unroll
    for (int ntl = 0; ntl < 4; ++ntl) {
      int nt = l * 4 + ntl;
      v8f acc = {};
      acc = wmma16(a0, ldb(w4p, nt, 0, 2, tl), acc);
      acc = wmma16(a1, ldb(w4p, nt, 1, 2, tl), acc);
      #pragma unroll
      for (int r = 0; r < 8; ++r)
        mixb[wv][(r + ((tl >> 4) << 3)) * CH + ntl * 16 + (tl & 15)] = acc[r];
    }
    if (valid) {
      if      (l == 0) emit_msgs<0>(sb, mb, half, vx, vy, vz, op);
      else if (l == 1) emit_msgs<1>(sb, mb, half, vx, vy, vz, op);
      else if (l == 2) emit_msgs<2>(sb, mb, half, vx, vy, vz, op);
      else             emit_msgs<3>(sb, mb, half, vx, vy, vz, op);
    }
  }
}

extern "C" void kernel_launch(void* const* d_in, const int* in_sizes, int n_in,
                              void* d_out, int out_size, void* d_ws, size_t ws_size,
                              hipStream_t stream) {
  const float* vectors    = (const float*)d_in[0];
  const float* node_feats = (const float*)d_in[1];
  const float* radial     = (const float*)d_in[2];
  const float* w1         = (const float*)d_in[3];
  const float* w2         = (const float*)d_in[4];
  const float* w3         = (const float*)d_in[5];
  const float* w4         = (const float*)d_in[6];
  const int*   senders    = (const int*)d_in[7];
  const int*   receivers  = (const int*)d_in[8];
  float*       out        = (float*)d_out;
  _Float16*    wsp        = (_Float16*)d_ws;

  const int E = in_sizes[0] / 3;

  // 1) zero the accumulation target
  zero_kernel<<<2048, 256, 0, stream>>>(out, out_size);
  // 2) pack + scale + f16-convert weights into WMMA B-fragment layout (26624 halves)
  pack_weights<<<104, 256, 0, stream>>>(w1, w2, w3, w4, wsp);
  // 3) main fused kernel: 4 waves/block, 16 edges/wave
  const int nTiles  = (E + 15) / 16;
  const int nBlocks = (nTiles + 3) / 4;
  mace_conv_kernel<<<nBlocks, 128, 0, stream>>>(vectors, node_feats, radial,
                                                senders, receivers, wsp, out, E);
}
